// Model_3332894622162
// MI455X (gfx1250) — matrix operand order repaired and checked by reading, not compiled, not run on hardware
//
#include <hip/hip_runtime.h>
#include <cstdint>
#include <cstddef>

// ---- problem constants (fixed by the reference harness) ----
#define RCUT    5.0f
#define NRAW    16
#define QQ      4
#define NSTRUCT 40
#define NATOMS  4000
#define FEAT    4096      // (L_MAX+1)*8*8*4*4
#define HID     256
#define MBLK1   64        // GEMM1 rows per block (4 M-frags)

typedef __bf16 bf16;
typedef __bf16 v16bf __attribute__((ext_vector_type(16)));
typedef __bf16 v8bf  __attribute__((ext_vector_type(8)));
typedef float  v8f   __attribute__((ext_vector_type(8)));
typedef int    v4i   __attribute__((ext_vector_type(4)));

// =====================================================================
// CDNA5 helpers
// =====================================================================
__device__ __forceinline__ v16bf pack16(v8bf lo, v8bf hi) {
  v16bf r;
#pragma unroll
  for (int t = 0; t < 8; ++t) { r[t] = lo[t]; r[t + 8] = hi[t]; }
  return r;
}

__device__ __forceinline__ v8f wmma_bf16(v16bf a, v16bf b, v8f c) {
  return __builtin_amdgcn_wmma_f32_16x16x32_bf16(false, a, false, b,
                                                 (short)0, c, false, false);
}

// Async 16B copy global -> LDS (CDNA5 GLOBAL_LOAD_ASYNC_TO_LDS_B128,
// tracked by ASYNCcnt). Builtin signature (from hipcc diagnostic):
//   (int4 addrspace(1)*, int4 addrspace(3)*, imm offset, imm cpol)
#if defined(__gfx1250__) && __has_builtin(__builtin_amdgcn_global_load_async_to_lds_b128)
#define HAVE_ASYNC_LDS 1
#else
#define HAVE_ASYNC_LDS 0
#endif

__device__ __forceinline__ void stage16(const bf16* g, bf16* l) {
#if HAVE_ASYNC_LDS
  __builtin_amdgcn_global_load_async_to_lds_b128(
      (__attribute__((address_space(1))) v4i*)g,
      (__attribute__((address_space(3))) v4i*)l, 0, 0);
#else
  *(v8bf*)l = *(const v8bf*)g;
#endif
}

__device__ __forceinline__ void stage_wait() {
#if HAVE_ASYNC_LDS
  asm volatile("s_wait_asynccnt 0" ::: "memory");
#endif
}

// =====================================================================
// Kernel 1: per-pair radial/angular expansion + atomic scatter into c
// c layout: [atom][q][lm][n]  (512 floats per atom)
// =====================================================================
__global__ void pair_scatter_kernel(const float* __restrict__ pos,
                                    const int*   __restrict__ pairs,
                                    const int*   __restrict__ species,
                                    const float* __restrict__ W_rad,   // [4][16][8]
                                    const float* __restrict__ W_alch,  // [4][4]
                                    float* __restrict__ c,
                                    int n_pairs) {
  int p = blockIdx.x * blockDim.x + threadIdx.x;
  if (p >= n_pairs) return;
  int ia = pairs[2 * p + 0];
  int ja = pairs[2 * p + 1];

  float dx = pos[3 * ja + 0] - pos[3 * ia + 0];
  float dy = pos[3 * ja + 1] - pos[3 * ia + 1];
  float dz = pos[3 * ja + 2] - pos[3 * ia + 2];
  float r  = sqrtf(dx * dx + dy * dy + dz * dz + 1e-12f);
  float inv = 1.0f / r;
  float x = dx * inv, y = dy * inv, z = dz * inv;

  float fcut = (r < RCUT) ? 0.5f * (cosf(3.14159265358979323846f * r / RCUT) + 1.0f) : 0.0f;

  const float sigma = RCUT / 16.0f;
  float R[4][8];
#pragma unroll
  for (int l = 0; l < 4; ++l)
#pragma unroll
    for (int n = 0; n < 8; ++n) R[l][n] = 0.0f;

#pragma unroll
  for (int k = 0; k < NRAW; ++k) {
    float mu = (RCUT / 15.0f) * (float)k;
    float d  = (r - mu) / sigma;
    float rk = __expf(-d * d) * fcut;
    const float* wr = W_rad + k * 8;
#pragma unroll
    for (int l = 0; l < 4; ++l)
#pragma unroll
      for (int n = 0; n < 8; ++n)
        R[l][n] = fmaf(rk, wr[l * 128 + n], R[l][n]);
  }

  float x2 = x * x, y2 = y * y, z2 = z * z;
  float Y[16];
  Y[0]  = 0.28209479177387814f;
  Y[1]  = 0.4886025119029199f * y;
  Y[2]  = 0.4886025119029199f * z;
  Y[3]  = 0.4886025119029199f * x;
  Y[4]  = 1.0925484305920792f * x * y;
  Y[5]  = 1.0925484305920792f * y * z;
  Y[6]  = 0.31539156525252005f * (3.0f * z2 - 1.0f);
  Y[7]  = 1.0925484305920792f * x * z;
  Y[8]  = 0.5462742152960396f * (x2 - y2);
  Y[9]  = 0.5900435899266435f * y * (3.0f * x2 - y2);
  Y[10] = 2.890611442640554f  * x * y * z;
  Y[11] = 0.4570457994644658f * y * (5.0f * z2 - 1.0f);
  Y[12] = 0.3731763325901154f * z * (5.0f * z2 - 3.0f);
  Y[13] = 0.4570457994644658f * x * (5.0f * z2 - 1.0f);
  Y[14] = 1.445305721320277f  * z * (x2 - y2);
  Y[15] = 0.5900435899266435f * x * (x2 - 3.0f * y2);

  int   sj = species[ja];
  float A0 = W_alch[sj * 4 + 0];
  float A1 = W_alch[sj * 4 + 1];
  float A2 = W_alch[sj * 4 + 2];
  float A3 = W_alch[sj * 4 + 3];

  float* cb = c + (size_t)ia * 512;
#pragma unroll
  for (int lm = 0; lm < 16; ++lm) {
    const int l = (lm >= 9) ? 3 : (lm >= 4) ? 2 : (lm >= 1) ? 1 : 0;
    float ylm = Y[lm];
#pragma unroll
    for (int n = 0; n < 8; ++n) {
      float t = ylm * R[l][n];
      atomicAdd(cb + 0 * 128 + lm * 8 + n, A0 * t);
      atomicAdd(cb + 1 * 128 + lm * 8 + n, A1 * t);
      atomicAdd(cb + 2 * 128 + lm * 8 + n, A2 * t);
      atomicAdd(cb + 3 * 128 + lm * 8 + n, A3 * t);
    }
  }
}

// =====================================================================
// Kernel 2: per-atom power spectrum + LayerNorm -> F (bf16, [atom][4096])
// =====================================================================
__global__ void atom_feature_kernel(const float* __restrict__ c,
                                    const float* __restrict__ gamma,
                                    const float* __restrict__ beta,
                                    bf16* __restrict__ F) {
  __shared__ float sc[512];
  __shared__ float sf[FEAT];
  __shared__ float red[256], red2[256];
  __shared__ float smean, srstd;

  int atom = blockIdx.x;
  int tid  = threadIdx.x;

  sc[tid]       = c[(size_t)atom * 512 + tid];
  sc[tid + 256] = c[(size_t)atom * 512 + tid + 256];
  __syncthreads();

  float ls = 0.0f, lq = 0.0f;
#pragma unroll
  for (int s = 0; s < 16; ++s) {
    int fi = tid + 256 * s;
    int l   = fi >> 10;
    int idx = fi & 1023;
    int a   = idx >> 8;
    int b   = (idx >> 6) & 3;
    int n   = (idx >> 3) & 7;
    int kk  = idx & 7;
    int lm0 = l * l;
    int nm  = 2 * l + 1;
    float acc = 0.0f;
    for (int m = 0; m < nm; ++m)
      acc = fmaf(sc[a * 128 + (lm0 + m) * 8 + n], sc[b * 128 + (lm0 + m) * 8 + kk], acc);
    acc *= rsqrtf((float)(2 * l + 1));
    sf[fi] = acc;
    ls += acc;
    lq = fmaf(acc, acc, lq);
  }

  red[tid] = ls; red2[tid] = lq;
  __syncthreads();
  for (int s = 128; s > 0; s >>= 1) {
    if (tid < s) { red[tid] += red[tid + s]; red2[tid] += red2[tid + s]; }
    __syncthreads();
  }
  if (tid == 0) {
    float mean = red[0] * (1.0f / FEAT);
    float var  = red2[0] * (1.0f / FEAT) - mean * mean;
    smean = mean;
    srstd = rsqrtf(var + 1e-5f);
  }
  __syncthreads();
  float mean = smean, rstd = srstd;

#pragma unroll
  for (int s = 0; s < 16; ++s) {
    int fi = tid + 256 * s;
    float v = (sf[fi] - mean) * rstd * gamma[fi] + beta[fi];
    F[(size_t)atom * FEAT + fi] = (bf16)v;
  }
}

// =====================================================================
// Weight pre-transpose + fp32->bf16: WT[q][n][k] = W[q][k][n]
// =====================================================================
__global__ void transpose_w1(const float* __restrict__ W1, bf16* __restrict__ W1T) {
  int idx = blockIdx.x * blockDim.x + threadIdx.x;
  if (idx >= QQ * FEAT * HID) return;
  int q = idx / (FEAT * HID);
  int r = idx % (FEAT * HID);
  int k = r / HID;
  int n = r % HID;
  W1T[((size_t)q * HID + n) * FEAT + k] = (bf16)W1[idx];
}

__global__ void transpose_w2(const float* __restrict__ W2, bf16* __restrict__ W2T) {
  int idx = blockIdx.x * blockDim.x + threadIdx.x;
  if (idx >= QQ * HID * HID) return;
  int q = idx / (HID * HID);
  int r = idx % (HID * HID);
  int k = r / HID;
  int n = r % HID;
  W2T[((size_t)q * HID + n) * HID + k] = (bf16)W2[idx];
}

// =====================================================================
// Kernel 4: GEMM1  G = F @ W1[q]; epilogue h = silu(s_{q,m}*G) -> bf16
// Block: 8 waves, tile 64(M) x 256(N). A tile (64x32 bf16, 4KB) staged
// in LDS via async copies; wave w owns N columns [w*32, w*32+32),
// 4 M-frags x 2 N-frags = 8 WMMA accumulators.
// =====================================================================
__global__ void gemm1_wmma(const bf16* __restrict__ F,
                           const bf16* __restrict__ W1T,
                           const int*  __restrict__ species,
                           const float* __restrict__ W_alch,
                           bf16* __restrict__ H1) {
  __shared__ bf16 sA[MBLK1 * 32];          // 4 KB

  const int q      = blockIdx.y;
  const int m_base = blockIdx.x * MBLK1;
  const int tid    = threadIdx.x;
  const int wave   = tid >> 5;
  const int lane   = tid & 31;
  const int row    = lane & 15;
  const bool hi    = lane >= 16;

  // B fragment sources: rows of W1T (one row = one output column)
  const bf16* Brow0 = W1T + ((size_t)q * HID + wave * 32 + row) * FEAT;
  const bf16* Brow1 = Brow0 + (size_t)16 * FEAT;
  const int aoff0 = hi ? 16 : 0;
  const int aoff1 = hi ? 24 : 8;
  const int boff  = hi ? 16 : 0;

  // A staging: 256 threads x 16B = 4KB; thread t copies row t>>2, chunk t&3
  int arow = m_base + (tid >> 2);
  if (arow > NATOMS - 1) arow = NATOMS - 1;
  const bf16* asrc = F + (size_t)arow * FEAT + (tid & 3) * 8;
  bf16* adst = sA + tid * 8;

  v8f acc[4][2];
#pragma unroll
  for (int mf = 0; mf < 4; ++mf)
#pragma unroll
    for (int nf = 0; nf < 2; ++nf)
      acc[mf][nf] = (v8f){0.f, 0.f, 0.f, 0.f, 0.f, 0.f, 0.f, 0.f};

  for (int k = 0; k < FEAT; k += 32) {
    if ((k & 511) == 0) {
      __builtin_prefetch(Brow0 + k + 512, 0, 3);   // global_prefetch_b8
      __builtin_prefetch(Brow1 + k + 512, 0, 3);
    }
    stage16(asrc + k, adst);                       // async global -> LDS
    stage_wait();                                  // s_wait_asynccnt 0
    __syncthreads();

    v16bf a[4];
#pragma unroll
    for (int mf = 0; mf < 4; ++mf) {
      const bf16* ar = sA + (mf * 16 + row) * 32;
      a[mf] = pack16(*(const v8bf*)(ar + aoff0), *(const v8bf*)(ar + aoff1));
    }
    v16bf b0 = pack16(*(const v8bf*)(Brow0 + k + boff),
                      *(const v8bf*)(Brow0 + k + boff + 8));
    v16bf b1 = pack16(*(const v8bf*)(Brow1 + k + boff),
                      *(const v8bf*)(Brow1 + k + boff + 8));
#pragma unroll
    for (int mf = 0; mf < 4; ++mf) {
      acc[mf][0] = wmma_bf16(a[mf], b0, acc[mf][0]);
      acc[mf][1] = wmma_bf16(a[mf], b1, acc[mf][1]);
    }
    __syncthreads();                               // protect sA reuse
  }

#pragma unroll
  for (int mf = 0; mf < 4; ++mf) {
#pragma unroll
    for (int r = 0; r < 8; ++r) {
      int m = m_base + mf * 16 + r + (hi ? 8 : 0);
      if (m < NATOMS) {
        float s = W_alch[species[m] * QQ + q];
#pragma unroll
        for (int nf = 0; nf < 2; ++nf) {
          float xv = s * acc[mf][nf][r];
          float yv = xv / (1.0f + __expf(-xv));    // silu
          H1[((size_t)q * NATOMS + m) * HID + wave * 32 + nf * 16 + row] = (bf16)yv;
        }
      }
    }
  }
}

// =====================================================================
// Kernel 5: GEMM2  h2 = silu(H1 @ W2[q]) -> bf16
// One wave computes a 32(M) x 64(N) tile: 2 x 4 = 8 accumulators, K=256.
// =====================================================================
__global__ void gemm2_wmma(const bf16* __restrict__ H1,
                           const bf16* __restrict__ W2T,
                           bf16* __restrict__ H2) {
  const int MT = NATOMS / 32;      // 125
  const int NT = HID / 64;         // 4
  int wave = blockIdx.x * (blockDim.x >> 5) + (threadIdx.x >> 5);
  if (wave >= QQ * MT * NT) return;
  int q   = wave / (MT * NT);
  int rem = wave % (MT * NT);
  int m0  = (rem / NT) * 32;
  int n0  = (rem % NT) * 64;

  int lane = threadIdx.x & 31;
  int row  = lane & 15;
  bool hi  = lane >= 16;
  const int aoff0 = hi ? 16 : 0;
  const int aoff1 = hi ? 24 : 8;
  const int boff  = hi ? 16 : 0;

  const bf16* Ar0 = H1 + ((size_t)q * NATOMS + m0 + row) * HID;
  const bf16* Ar1 = Ar0 + (size_t)16 * HID;

  v8f acc[2][4];
#pragma unroll
  for (int mf = 0; mf < 2; ++mf)
#pragma unroll
    for (int nf = 0; nf < 4; ++nf)
      acc[mf][nf] = (v8f){0.f, 0.f, 0.f, 0.f, 0.f, 0.f, 0.f, 0.f};

#pragma unroll
  for (int k = 0; k < HID; k += 32) {
    v16bf a0 = pack16(*(const v8bf*)(Ar0 + k + aoff0),
                      *(const v8bf*)(Ar0 + k + aoff1));
    v16bf a1 = pack16(*(const v8bf*)(Ar1 + k + aoff0),
                      *(const v8bf*)(Ar1 + k + aoff1));
#pragma unroll
    for (int nf = 0; nf < 4; ++nf) {
      const bf16* Br = W2T + ((size_t)q * HID + n0 + nf * 16 + row) * HID;
      v16bf b = pack16(*(const v8bf*)(Br + k + boff),
                       *(const v8bf*)(Br + k + boff + 8));
      acc[0][nf] = wmma_bf16(a0, b, acc[0][nf]);
      acc[1][nf] = wmma_bf16(a1, b, acc[1][nf]);
    }
  }

#pragma unroll
  for (int mf = 0; mf < 2; ++mf) {
#pragma unroll
    for (int r = 0; r < 8; ++r) {
      int m = m0 + mf * 16 + r + (hi ? 8 : 0);
#pragma unroll
      for (int nf = 0; nf < 4; ++nf) {
        float xv = acc[mf][nf][r];
        float yv = xv / (1.0f + __expf(-xv));
        H2[((size_t)q * NATOMS + m) * HID + n0 + nf * 16 + row] = (bf16)yv;
      }
    }
  }
}

// =====================================================================
// Kernel 6: last layer + composition + segment sum over structures
// =====================================================================
__global__ void final_reduce(const bf16* __restrict__ H2,
                             const float* __restrict__ W_last,   // [Q][256][1]
                             const float* __restrict__ comp_w,
                             const int*  __restrict__ species,
                             const int*  __restrict__ sid,
                             float* __restrict__ out) {
  int n = blockIdx.x * blockDim.x + threadIdx.x;
  if (n >= NATOMS) return;
  float e = 0.0f;
#pragma unroll
  for (int q = 0; q < QQ; ++q) {
    const bf16*  hr = H2 + ((size_t)q * NATOMS + n) * HID;
    const float* wl = W_last + q * HID;
    float acc = 0.0f;
    for (int h = 0; h < HID; ++h) acc = fmaf((float)hr[h], wl[h], acc);
    e += acc;
  }
  float val = (e + comp_w[species[n]]) * 0.5f;     // / sqrt(Q=4)
  atomicAdd(&out[sid[n]], val);
}

// =====================================================================
// Host-side launcher
// =====================================================================
extern "C" void kernel_launch(void* const* d_in, const int* in_sizes, int n_in,
                              void* d_out, int out_size, void* d_ws, size_t ws_size,
                              hipStream_t stream) {
  const float* positions = (const float*)d_in[0];
  const int*   pairs     = (const int*)  d_in[1];
  const int*   species   = (const int*)  d_in[2];
  const int*   sids      = (const int*)  d_in[3];
  const float* W_rad     = (const float*)d_in[4];
  const float* W_alch    = (const float*)d_in[5];
  const float* gamma     = (const float*)d_in[6];
  const float* beta      = (const float*)d_in[7];
  const float* W1        = (const float*)d_in[8];
  const float* W2        = (const float*)d_in[9];
  const float* W_last    = (const float*)d_in[10];
  const float* comp_w    = (const float*)d_in[11];
  float*       out       = (float*)d_out;

  int n_pairs = in_sizes[1] / 2;

  char* w = (char*)d_ws;
  auto carve = [&](size_t bytes) -> char* {
    char* p = w;
    w += (bytes + 255) & ~(size_t)255;
    return p;
  };
  float* c   = (float*)carve((size_t)NATOMS * 512 * sizeof(float));
  bf16*  F   = (bf16*) carve((size_t)NATOMS * FEAT * sizeof(bf16));
  bf16*  W1T = (bf16*) carve((size_t)QQ * HID * FEAT * sizeof(bf16));
  bf16*  W2T = (bf16*) carve((size_t)QQ * HID * HID * sizeof(bf16));
  bf16*  H1  = (bf16*) carve((size_t)QQ * NATOMS * HID * sizeof(bf16));
  bf16*  H2  = (bf16*) carve((size_t)QQ * NATOMS * HID * sizeof(bf16));

  (void)hipMemsetAsync(c, 0, (size_t)NATOMS * 512 * sizeof(float), stream);
  (void)hipMemsetAsync(out, 0, NSTRUCT * sizeof(float), stream);

  pair_scatter_kernel<<<(n_pairs + 255) / 256, 256, 0, stream>>>(
      positions, pairs, species, W_rad, W_alch, c, n_pairs);

  transpose_w1<<<(QQ * FEAT * HID + 255) / 256, 256, 0, stream>>>(W1, W1T);
  transpose_w2<<<(QQ * HID * HID + 255) / 256, 256, 0, stream>>>(W2, W2T);

  atom_feature_kernel<<<NATOMS, 256, 0, stream>>>(c, gamma, beta, F);

  dim3 g1((NATOMS + MBLK1 - 1) / MBLK1, QQ);       // 63 x 4 blocks
  gemm1_wmma<<<g1, 256, 0, stream>>>(F, W1T, species, W_alch, H1);

  const int tiles2 = QQ * (NATOMS / 32) * (HID / 64);   // 2000 waves
  gemm2_wmma<<<(tiles2 + 7) / 8, 256, 0, stream>>>(H1, W2T, H2);

  final_reduce<<<(NATOMS + 255) / 256, 256, 0, stream>>>(
      H2, W_last, comp_w, species, sids, out);
}
